// GAE1_34737695490299
// MI455X (gfx1250) — matrix operand order repaired and checked by reading, not compiled, not run on hardware
//
#include <hip/hip_runtime.h>

#define DEV_INLINE __device__ __forceinline__

typedef __attribute__((ext_vector_type(8)))  unsigned short u16x8;
typedef __attribute__((ext_vector_type(16))) __bf16        v16bf;
typedef __attribute__((ext_vector_type(2)))  __bf16        bf16x2;
typedef __attribute__((ext_vector_type(8)))  float         v8f;
typedef __attribute__((ext_vector_type(2)))  float         f32x2;
typedef __attribute__((ext_vector_type(4)))  float         f32x4;
typedef __attribute__((ext_vector_type(4)))  unsigned int  u32x4;
typedef __attribute__((ext_vector_type(8)))  int           i32x8;
typedef __attribute__((ext_vector_type(4)))  int           i32x4;

// Assemble a WMMA bf16 operand (16 x bf16 = 8 VGPRs) from two 16B LDS loads.
union FragCvt { struct { u16x8 lo, hi; } u; v16bf v; };

// A-matrix 16x32 bf16 fragment (CDNA5 ISA layout): lane L: M=L&15, hi=L>>4;
// elements 0..7 = row[8*hi..], elements 8..15 = row[16+8*hi..].
DEV_INLINE v16bf ld_a_frag(const __bf16* row, int hi) {
  FragCvt f;
  f.u.lo = *(const u16x8*)(row + 8 * hi);
  f.u.hi = *(const u16x8*)(row + 16 + 8 * hi);
  return f.v;
}

// B-matrix 32x16 bf16 fragment: lane L: N=L&15, hi=L>>4;
// elements 0..7 = col[8*hi..], elements 8..15 = col[16+8*hi..].
DEV_INLINE v16bf ld_b_frag(const __bf16* col, int hi) {
  FragCvt f;
  f.u.lo = *(const u16x8*)(col + 8 * hi);
  f.u.hi = *(const u16x8*)(col + 16 + 8 * hi);
  return f.v;
}

// ---------------------------------------------------------------------------
// Tensor Data Mover: async 2-D tile load adj[32][32] (int32) -> LDS.
// D# packing per CDNA5 ISA (§8.3 group0, §8.4 group1):
//   g0: count=1 | lds_addr | global_addr[56:0] | type=2
//   g1: data_size=4B, tensor_dim0=tensor_dim1=8192, tile 32x32,
//       tensor_dim0_stride=8192 elements, no multicast/pad/iterate.
// Tracked with TENSORcnt (s_wait_tensorcnt). 6-arg builtin (clang-23 form).
// ---------------------------------------------------------------------------
DEV_INLINE void tdm_load_adj(const int* gptr, unsigned ldsOff) {
  const unsigned long long ga = (unsigned long long)gptr;
  u32x4 g0;
  g0[0] = 1u;                                                  // 1 valid D#
  g0[1] = ldsOff;                                              // LDS byte addr
  g0[2] = (unsigned)ga;                                        // gaddr[31:0]
  g0[3] = (unsigned)((ga >> 32) & 0x01ffffffu) | 0x80000000u;  // [56:32]|type2
  i32x8 g1;
  g1[0] = 2 << 16;      // workgroup_mask=0, data_size=2 (4 bytes)
  g1[1] = 0x20000000;   // tensor_dim0 = 8192 (bits 79:48, low16 here)
  g1[2] = 0x20000000;   // tensor_dim0 hi16 = 0, tensor_dim1 = 8192 (low16)
  g1[3] = 32 << 16;     // tensor_dim1 hi16 = 0, tile_dim0 = 32
  g1[4] = 32;           // tile_dim1 = 32, tile_dim2 = 0
  g1[5] = 8192;         // tensor_dim0_stride[31:0] = 8192 elements
  g1[6] = 0;            // stride hi, tensor_dim1_stride lo
  g1[7] = 0;
  const i32x4 z4 = (i32x4)0;
  const i32x8 z8 = (i32x8)0;
  __builtin_amdgcn_tensor_load_to_lds(g0, g1, z4, z4, z8, 0);
}

// ---------------------------------------------------------------------------
// h = in(fp32)[N,FIN] @ w(fp32)[FIN,FOUT]  ->  hf (fp32) and hb (bf16)
// ---------------------------------------------------------------------------
template <int FIN, int FOUT>
__global__ __launch_bounds__(2 * (FOUT / 64) * 32)
void k_gemm(const float* __restrict__ in, const float* __restrict__ wmat,
            float* __restrict__ hf, __bf16* __restrict__ hb) {
  constexpr int NW  = 2 * (FOUT / 64);
  constexpr int BLK = NW * 32;
  __shared__ __bf16 At[32][40];    // [row][k], 80B row stride
  __shared__ __bf16 Wt[FOUT][40];  // [col][k], transposed for B-frag runs
  const int tid = threadIdx.x, lane = tid & 31, w = tid >> 5;
  const int m = lane & 15, hi = lane >> 4;
  const int rt = w & 1, cg = w >> 1;
  const int rowBase = blockIdx.x * 32;
  v8f acc[4] = {};
  for (int kt = 0; kt < FIN / 32; ++kt) {
    const int k0 = kt * 32;
    __syncthreads();
    // A tile: 32x32 fp32 -> bf16 pairs (exact trip counts, no exec masking)
#pragma unroll
    for (int it = 0; it < 512 / BLK; ++it) {
      const int idx = tid + it * BLK;
      const int r = idx >> 4, kp = (idx & 15) * 2;
      const f32x2 fv = *(const f32x2*)&in[(size_t)(rowBase + r) * FIN + k0 + kp];
      bf16x2 bv;
      bv.x = (__bf16)fv.x;
      bv.y = (__bf16)fv.y;
      *(bf16x2*)&At[r][kp] = bv;
    }
    // W tile: wmat[k][c] -> Wt[c][k], 16B global reads
#pragma unroll
    for (int it = 0; it < (8 * FOUT) / BLK; ++it) {
      const int idx = tid + it * BLK;
      const int k = (idx * 4) / FOUT, c0 = (idx * 4) % FOUT;
      const f32x4 fv = *(const f32x4*)&wmat[(size_t)(k0 + k) * FOUT + c0];
#pragma unroll
      for (int u = 0; u < 4; ++u) Wt[c0 + u][k] = (__bf16)fv[u];
    }
    __syncthreads();
    // load all fragments first (distinct regs) so DS loads overlap, then
    // issue the 4 WMMAs back-to-back
    const v16bf af = ld_a_frag(&At[rt * 16 + m][0], hi);
    v16bf bfr[4];
#pragma unroll
    for (int t = 0; t < 4; ++t)
      bfr[t] = ld_b_frag(&Wt[cg * 64 + t * 16 + m][0], hi);
#pragma unroll
    for (int t = 0; t < 4; ++t)
      acc[t] = __builtin_amdgcn_wmma_f32_16x16x32_bf16(false, af, false, bfr[t],
                                                       (short)0, acc[t], false, false);
  }
#pragma unroll
  for (int t = 0; t < 4; ++t) {
    const int c = cg * 64 + t * 16 + m;
#pragma unroll
    for (int v = 0; v < 8; ++v) {
      const size_t r = (size_t)rowBase + rt * 16 + 8 * hi + v;
      const float val = acc[t][v];
      hf[r * FOUT + c] = val;
      hb[r * FOUT + c] = (__bf16)val;
    }
  }
}

// ---------------------------------------------------------------------------
// src_i = h_i . a[0:FOUT],  tgt_i = h_i . a[FOUT:2*FOUT]   (one wave per row)
// ---------------------------------------------------------------------------
template <int FOUT>
__global__ void k_src_tgt(const float* __restrict__ h, const float* __restrict__ a,
                          float* __restrict__ src, float* __restrict__ tgt) {
  const int lane = threadIdx.x & 31;
  const int row  = (int)((blockIdx.x * blockDim.x + threadIdx.x) >> 5);
  const float* hr = h + (size_t)row * FOUT;
  float s = 0.f, t = 0.f;
#pragma unroll
  for (int c = lane; c < FOUT; c += 32) {
    const float hv = hr[c];
    s += hv * a[c];
    t += hv * a[FOUT + c];
  }
#pragma unroll
  for (int off = 16; off; off >>= 1) {
    s += __shfl_xor(s, off, 32);
    t += __shfl_xor(t, off, 32);
  }
  if (lane == 0) { src[row] = s; tgt[row] = t; }
}

__global__ void k_tgt_max(const float* __restrict__ tgt, float* __restrict__ tmax) {
  __shared__ float red[1024];
  float mv = -3.4e38f;
  for (int i = threadIdx.x; i < 8192; i += 1024) mv = fmaxf(mv, tgt[i]);
  red[threadIdx.x] = mv;
  __syncthreads();
  for (int s = 512; s > 0; s >>= 1) {
    if ((int)threadIdx.x < s) red[threadIdx.x] = fmaxf(red[threadIdx.x], red[threadIdx.x + s]);
    __syncthreads();
  }
  if (threadIdx.x == 0) tmax[0] = red[0];
}

// ---------------------------------------------------------------------------
// Fused masked softmax-attention:  out = softmax(mask(leaky(src+tgt^T))) @ h
// m_i = leaky(src_i + max_j tgt_j) >= true masked row max (leaky monotone),
// so exp(e - m_i) <= 1: exact softmax in one pass, no online rescaling.
// adj tiles stream from HBM via TDM (double-buffered), overlapped with WMMA.
// ---------------------------------------------------------------------------
template <int FOUT>
__global__ __launch_bounds__(2 * (FOUT / 64) * 32)
void k_attn(const float* __restrict__ src, const float* __restrict__ tgt,
            const float* __restrict__ tmax, const int* __restrict__ adj,
            const __bf16* __restrict__ hb, float* __restrict__ out) {
  constexpr int NW  = 2 * (FOUT / 64);
  constexpr int BLK = NW * 32;
  constexpr int RPW = 32 / NW;     // score rows owned per wave
  constexpr int KT  = 8192 / 32;   // column tiles
  __shared__ __bf16 Pt[32][40];    // P tile [row][j] bf16
  __shared__ __bf16 Ht[FOUT][40];  // H tile transposed [col][k] bf16
  __shared__ int    AdjT[2][32][32];  // TDM double buffer
  __shared__ float  ls[32];
  const int tid = threadIdx.x, lane = tid & 31, w = tid >> 5;
  const int m = lane & 15, hi = lane >> 4;
  const int rt = w & 1, cg = w >> 1;
  const int rowBase = blockIdx.x * 32;

  // loop-invariant per-row state in registers
  float sarr[RPW], marr[RPW], lpart[RPW];
  const float tm = tmax[0];
#pragma unroll
  for (int q = 0; q < RPW; ++q) {
    const float s = src[rowBase + w * RPW + q];
    const float e = s + tm;
    sarr[q]  = s;
    marr[q]  = e > 0.f ? e : 0.2f * e;
    lpart[q] = 0.f;
  }
  if (w == 0)  // kick off adj tile 0
    tdm_load_adj(adj + (size_t)rowBase * 8192,
                 (unsigned)(size_t)(void*)&AdjT[0][0][0]);

  v8f acc[4] = {};
  for (int kt = 0; kt < KT; ++kt) {
    const int j0 = kt * 32;
    const float tj = tgt[j0 + lane];
    __syncthreads();  // (A) previous tile's LDS reads complete
    if (w == 0) {
      if (kt + 1 < KT) {  // prefetch next adj tile, then ensure current landed
        tdm_load_adj(adj + (size_t)rowBase * 8192 + (j0 + 32),
                     (unsigned)(size_t)(void*)&AdjT[(kt + 1) & 1][0][0]);
        __builtin_amdgcn_s_wait_tensorcnt(1);
      } else {
        __builtin_amdgcn_s_wait_tensorcnt(0);
      }
    }
    // ---- stage H tile transposed (overlaps wave0's tensorcnt wait) ----
    const u16x8* hsrc = (const u16x8*)(hb + (size_t)j0 * FOUT);
#pragma unroll
    for (int it = 0; it < 4; ++it) {   // 32*FOUT/8 elems, BLK==FOUT threads
      const int idx = tid + it * BLK;
      const u16x8 vv = hsrc[idx];
      const int k = (idx * 8) / FOUT, c = (idx * 8) % FOUT;
#pragma unroll
      for (int u = 0; u < 8; ++u) Ht[c + u][k] = ((const __bf16*)&vv)[u];
    }
    if (kt + 1 < KT)  // gfx1250 global_prefetch_b8 for the next H tile
      __builtin_prefetch(hb + (size_t)(j0 + 32) * FOUT + (size_t)tid * 8, 0, 1);
    __syncthreads();  // (B) AdjT[kt&1] + Ht visible to all waves
    // ---- scores from LDS adj: p = adj ? exp(leaky(src+tgt) - m) : 0 ----
    const int* arow = &AdjT[kt & 1][w * RPW][lane];
#pragma unroll
    for (int q = 0; q < RPW; ++q) {
      const int av = arow[q * 32];
      float e = sarr[q] + tj;
      e = e > 0.f ? e : 0.2f * e;
      const float p = (av > 0) ? __expf(e - marr[q]) : 0.f;
      Pt[w * RPW + q][lane] = (__bf16)p;
      lpart[q] += p;   // per-lane partial; reduced once after the loop
    }
    __syncthreads();  // (C) Pt visible
    // ---- acc += P(16x32) x H(32x64 per wave): load all frags, then WMMA ----
    const v16bf af = ld_a_frag(&Pt[rt * 16 + m][0], hi);
    v16bf bfr[4];
#pragma unroll
    for (int t = 0; t < 4; ++t)
      bfr[t] = ld_b_frag(&Ht[cg * 64 + t * 16 + m][0], hi);
#pragma unroll
    for (int t = 0; t < 4; ++t)
      acc[t] = __builtin_amdgcn_wmma_f32_16x16x32_bf16(false, af, false, bfr[t],
                                                       (short)0, acc[t], false, false);
  }
  // single final reduction of the softmax denominators
#pragma unroll
  for (int q = 0; q < RPW; ++q) {
    float ps = lpart[q];
#pragma unroll
    for (int off = 16; off; off >>= 1) ps += __shfl_xor(ps, off, 32);
    if (lane == 0) ls[w * RPW + q] = ps;
  }
  __syncthreads();
  float linv[8];
#pragma unroll
  for (int v = 0; v < 8; ++v)
    linv[v] = 1.0f / fmaxf(ls[rt * 16 + 8 * hi + v], 1e-30f);
#pragma unroll
  for (int t = 0; t < 4; ++t) {
    const int c = cg * 64 + t * 16 + m;
#pragma unroll
    for (int v = 0; v < 8; ++v) {
      const size_t r = (size_t)rowBase + rt * 16 + 8 * hi + v;
      out[r * FOUT + c] = acc[t][v] * linv[v];
    }
  }
}

// ---------------------------------------------------------------------------
template <int FIN, int FOUT>
static void run_layer(const float* in, const float* wmat, const float* avec,
                      const int* adj, float* hf, __bf16* hb,
                      float* src, float* tgt, float* tmax,
                      float* out, hipStream_t s) {
  k_gemm<FIN, FOUT><<<256, 2 * (FOUT / 64) * 32, 0, s>>>(in, wmat, hf, hb);
  k_src_tgt<FOUT><<<1024, 256, 0, s>>>(hf, avec, src, tgt);
  k_tgt_max<<<1, 1024, 0, s>>>(tgt, tmax);
  k_attn<FOUT><<<256, 2 * (FOUT / 64) * 32, 0, s>>>(src, tgt, tmax, adj, hb, out);
}

extern "C" void kernel_launch(void* const* d_in, const int* in_sizes, int n_in,
                              void* d_out, int out_size, void* d_ws, size_t ws_size,
                              hipStream_t stream) {
  (void)in_sizes; (void)n_in; (void)out_size; (void)ws_size;
  constexpr int N = 8192;
  const float* x   = (const float*)d_in[0];
  const int*   adj = (const int*)  d_in[1];
  const float* w1  = (const float*)d_in[2]; const float* a1 = (const float*)d_in[3];
  const float* w2  = (const float*)d_in[4]; const float* a2 = (const float*)d_in[5];
  const float* w3  = (const float*)d_in[6]; const float* a3 = (const float*)d_in[7];
  const float* w4  = (const float*)d_in[8]; const float* a4 = (const float*)d_in[9];

  float* xbar = (float*)d_out;                    // [N,256] = layer-4 output
  float* h2   = (float*)d_out + (size_t)N * 256;  // [N, 64] = layer-2 output

  // workspace carve-out (~16.2 MB)
  char* p = (char*)d_ws;
  float*  hf    = (float*)p;  p += (size_t)N * 256 * sizeof(float);
  __bf16* hb    = (__bf16*)p; p += (size_t)N * 256 * sizeof(__bf16);
  float*  inter = (float*)p;  p += (size_t)N * 128 * sizeof(float);
  float*  src   = (float*)p;  p += (size_t)N * sizeof(float);
  float*  tgt   = (float*)p;  p += (size_t)N * sizeof(float);
  float*  tmax  = (float*)p;

  run_layer<256, 128>(x,     w1, a1, adj, hf, hb, src, tgt, tmax, inter, stream);
  run_layer<128,  64>(inter, w2, a2, adj, hf, hb, src, tgt, tmax, h2,    stream);
  run_layer< 64, 128>(h2,    w3, a3, adj, hf, hb, src, tgt, tmax, inter, stream);
  run_layer<128, 256>(inter, w4, a4, adj, hf, hb, src, tgt, tmax, xbar,  stream);
}
